// WaterNet1104_16690242913228
// MI455X (gfx1250) — matrix operand order repaired and checked by reading, not compiled, not run on hardware
//
#include <hip/hip_runtime.h>
#include <hip/hip_bf16.h>

typedef __attribute__((ext_vector_type(16))) _Float16 v16h;
typedef __attribute__((ext_vector_type(8)))  float    v8f;

#define NHH 64
#define NSS 1024
#define NTT 365
#define NGG 32
#define HID 256
#define NVV 192   /* 3*nh */
#define NWW 576   /* 9*nh */

#define TS 16          /* sites per block (WMMA M) */
#define THREADS 384    /* 12 waves: one 16-wide N-tile per wave */
#define PITCH_H 264    /* half pitch: 132 dw = 4 mod 64 banks */
#define PITCH_V 196    /* float pitch: 4 mod 64 banks */

__device__ __forceinline__ float sigf(float x) { return 1.0f / (1.0f + __expf(-x)); }
// fast sigmoid / tanh: 1-ulp v_rcp; result is quantized (f16 or fed to bounded gates)
__device__ __forceinline__ float sig_fast(float x) {
  return __builtin_amdgcn_rcpf(1.0f + __expf(-x));
}
__device__ __forceinline__ float tanh_fast(float x) {
  return fmaf(-2.0f, __builtin_amdgcn_rcpf(1.0f + __expf(2.0f * x)), 1.0f);
}

// hb[s][j] = sum_k xc[s][k] * WT1[k+1][j] + bT1[j]   (layer-1 constant part, f32 exact)
__global__ void prep_hb(const float* __restrict__ xc, const float* __restrict__ WT1,
                        const float* __restrict__ bT1, float* __restrict__ hb) {
  int s = blockIdx.x, j = threadIdx.x;
  const float* xr = xc + s * NGG;
  float acc = bT1[j];
#pragma unroll
  for (int k = 0; k < NGG; ++k) acc = fmaf(xr[k], WT1[(k + 1) * HID + j], acc);
  hb[s * HID + j] = acc;
}

// static gate net: w = tanh(xc@W1+b1)@W2+b2 -> 9 processed gate arrays (f32 exact)
__global__ void gates_kernel(const float* __restrict__ xc,
                             const float* __restrict__ W1, const float* __restrict__ b1,
                             const float* __restrict__ W2, const float* __restrict__ b2,
                             float* __restrict__ gates) {
  __shared__ float sh1[HID];
  __shared__ float sw6[NHH];
  int s = blockIdx.x, t = threadIdx.x;
  const float* xr = xc + s * NGG;
  for (int j = t; j < HID; j += NHH) {
    float a = b1[j];
#pragma unroll
    for (int k = 0; k < NGG; ++k) a = fmaf(xr[k], W1[k * HID + j], a);
    sh1[j] = tanhf(a);
  }
  __syncthreads();
  float w[9];
#pragma unroll
  for (int c = 0; c < 9; ++c) {
    float a = b2[c * NHH + t];
    for (int k = 0; k < HID; ++k) a = fmaf(sh1[k], W2[k * NWW + c * NHH + t], a);
    w[c] = a;
  }
  sw6[t] = w[6];
  __syncthreads();
  float m = -3.4e38f;
  for (int i = 0; i < NHH; ++i) m = fmaxf(m, sw6[i]);
  float ssum = 0.0f;
  for (int i = 0; i < NHH; ++i) ssum += __expf(sw6[i] - m);
  int o = s * NHH + t;
  gates[0 * NSS * NHH + o] = __expf(w[0]) + 1.0f;        // gm
  gates[1 * NSS * NHH + o] = sigf(w[1]);                 // k1
  gates[2 * NSS * NHH + o] = sigf(w[2]);                 // k2
  gates[3 * NSS * NHH + o] = sigf(w[3]);                 // k23
  gates[4 * NSS * NHH + o] = sigf(w[4]) * 0.1f;          // k3
  gates[5 * NSS * NHH + o] = __expf(w[5]) * 2.0f;        // gl
  gates[6 * NSS * NHH + o] = __expf(w[6] - m) / ssum;    // ga (softmax over nh)
  gates[7 * NSS * NHH + o] = fmaxf(w[7], 0.0f);          // qb
  gates[8 * NSS * NHH + o] = sigf(w[8]) * 5.0f;          // ge
}

// Fused time scan: per step h=tanh(hb + LAI*WT1row0) (f16), v = h @ WT2 via WMMA
// (B fragments register-resident across all 365 steps), sigmoids, bucket update, Y.
__global__ __launch_bounds__(THREADS) void scan_kernel(
    const float* __restrict__ x, const float* __restrict__ WT1,
    const float* __restrict__ WT2, const float* __restrict__ bT2,
    const float* __restrict__ hb, const float* __restrict__ gates,
    float* __restrict__ out) {
  extern __shared__ char smem[];
  _Float16* h16 = (_Float16*)smem;                  // [16][264] halfs
  float* vsm = (float*)(h16 + TS * PITCH_H);        // [16][196] floats (vi|vk|ve)
  float* g_s = vsm + TS * PITCH_V;                  // [9][16*64]
  float* w10 = g_s + 9 * TS * NHH;                  // [256] WT1 row 0 (LAI row)
  float* met = w10 + HID;                           // [5][16]: Ps, Pl, Ta, E, LAI

  int tid = threadIdx.x;
  int s0 = blockIdx.x * TS;
  int lane = tid & 31;
  int wave = tid >> 5;          // 0..11 -> N-tile = wave*16
  int m = lane & 15;
  int lh = lane >> 4;           // half-wave select

  for (int c = tid; c < HID; c += THREADS) w10[c] = WT1[c];
  for (int idx = tid; idx < 9 * TS * NHH; idx += THREADS) {
    int g = idx >> 10, r = idx & (TS * NHH - 1);
    g_s[idx] = gates[g * NSS * NHH + s0 * NHH + r];
  }

  // ---- register-resident B fragments for this wave's 16-column tile ----
  int nbw = wave * 16;
  float biasr = bT2[nbw + m];
  float scale = (wave >= 8) ? 5.0f : 1.0f;   // ve tiles get *5
  v16h bfr[8];
#pragma unroll
  for (int kk = 0; kk < 8; ++kk) {
    union { v16h v; _Float16 e[16]; } u;
    int kbase = kk * 32 + lh * 16;
#pragma unroll
    for (int i = 0; i < 16; ++i)
      u.e[i] = (_Float16)WT2[(kbase + i) * NVV + nbw + m];
    bfr[kk] = u.v;
  }

  float S0[4], Sv[4], S2s[4], S3[4];
#pragma unroll
  for (int j = 0; j < 4; ++j) { S0[j] = 0.f; Sv[j] = 0.f; S2s[j] = 0.f; S3[j] = 0.f; }
  int sl = tid >> 4;             // site owned for the state update (tid<256)
  int h0 = (tid & 15) << 2;      // 4 hidden units owned
  __syncthreads();

  for (int t = 0; t < NTT; ++t) {
    // ---- forcings & rain/snow partition (16 lanes) ----
    if (tid < TS) {
      const float* xp = x + ((size_t)t * NSS + s0 + tid) * 5;
      float P = xp[0], E = xp[1], T1 = xp[2], T2 = xp[3], LAI = xp[4];
      bool inter = (T1 < 0.f) && (T2 > 0.f);
      float denom = inter ? (T2 - T1) : 1.0f;
      float ratio = inter ? fminf(1.f, fmaxf(-1.f, (T1 + T2) / denom)) : 0.f;
      float rP = 1.0f - acosf(ratio) * (1.0f / 3.1415f);
      if (T1 >= 0.f) rP = 1.0f;
      if (T2 <= 0.f) rP = 0.0f;
      met[tid]      = P * (1.0f - rP);   // Ps
      met[16 + tid] = P * rP;            // Pl
      met[32 + tid] = 0.5f * (T1 + T2);  // Ta
      met[48 + tid] = E;
      met[64 + tid] = LAI;
    }
    __syncthreads();
    // ---- h = tanh(hb + LAI * WT1row0), stored f16 for WMMA A ----
    for (int idx = tid; idx < TS * HID; idx += THREADS) {
      int c = idx & (HID - 1), r = idx >> 8;
      h16[r * PITCH_H + c] =
          (_Float16)tanh_fast(fmaf(met[64 + r], w10[c], hb[(s0 + r) * HID + c]));
    }
    __syncthreads();
    // ---- GEMM: one 16x16 tile per wave; only A comes from LDS ----
    v8f acc = {};
    const _Float16* arow = h16 + m * PITCH_H + lh * 16;
#pragma unroll
    for (int kk = 0; kk < 8; ++kk) {
      v16h af;
      ((float4*)&af)[0] = *(const float4*)(arow + kk * 32);
      ((float4*)&af)[1] = *(const float4*)(arow + kk * 32 + 8);
      acc = __builtin_amdgcn_wmma_f32_16x16x32_f16(
          false, af, false, bfr[kk], (short)0, acc, false, false);
    }
#pragma unroll
    for (int r = 0; r < 8; ++r) {
      int M = r + lh * 8;
      vsm[M * PITCH_V + nbw + m] = scale * sig_fast(acc[r] + biasr);
    }
    __syncthreads();
    // ---- state update (threads 0..255: fixed site, 4 hidden units) ----
    if (tid < 256) {
      float Ps = met[sl], Pl = met[16 + sl], Ta = met[32 + sl], Ee = met[48 + sl];
      float ysum = 0.0f;
#pragma unroll
      for (int j = 0; j < 4; ++j) {
        int h = h0 + j, gi = sl * NHH + h;
        float vi = vsm[sl * PITCH_V + h];
        float vk = vsm[sl * PITCH_V + NHH + h];
        float ve = vsm[sl * PITCH_V + 2 * NHH + h];
        float gm  = g_s[0 * TS * NHH + gi];
        float k1  = g_s[1 * TS * NHH + gi];
        float k2  = g_s[2 * TS * NHH + gi];
        float k23 = g_s[3 * TS * NHH + gi];
        float k3  = g_s[4 * TS * NHH + gi];
        float gl  = g_s[5 * TS * NHH + gi];
        float ga  = g_s[6 * TS * NHH + gi];
        float qb  = g_s[7 * TS * NHH + gi];
        float ge  = g_s[8 * TS * NHH + gi];
        float Pl1 = Pl * (1.0f - vi), Pl2 = Pl * vi, Ev = Ee * ve;
        float H0 = S0[j] + Ps;
        float qSm = fminf(H0, fmaxf(Ta * gm, 0.0f));
        float Hv = fmaxf(Sv[j] + Pl1 - Ev, 0.0f);
        float qv = Hv * vk;
        float H2 = fmaxf(S2s[j] + qSm + qv - Ee * ge + Pl2, 0.0f);
        float base = fmaxf(H2 - gl, 0.0f);
        float Q1 = (base > 0.0f) ? __expf(k1 * __logf(base)) : 0.0f;
        float q2 = fminf(H2, gl) * k2;
        float Q2 = q2 * (1.0f - k23);
        float H3 = S3[j] + q2 * k23;
        float Q3 = H3 * k3 + qb;
        S0[j] = H0 - qSm;
        Sv[j] = Hv - qv;
        S2s[j] = H2 - Q1 - q2;
        S3[j] = H3 - Q3;
        ysum += (Q1 + Q2 + Q3) * ga;
      }
      ysum += __shfl_down(ysum, 8, 16);
      ysum += __shfl_down(ysum, 4, 16);
      ysum += __shfl_down(ysum, 2, 16);
      ysum += __shfl_down(ysum, 1, 16);
      if ((tid & 15) == 0) out[(size_t)t * NSS + s0 + sl] = ysum;
    }
    __syncthreads();
  }
}

extern "C" void kernel_launch(void* const* d_in, const int* in_sizes, int n_in,
                              void* d_out, int out_size, void* d_ws, size_t ws_size,
                              hipStream_t stream) {
  (void)in_sizes; (void)n_in; (void)out_size; (void)ws_size;
  const float* x   = (const float*)d_in[0];
  const float* xc  = (const float*)d_in[1];
  const float* W1  = (const float*)d_in[2];
  const float* b1  = (const float*)d_in[3];
  const float* W2  = (const float*)d_in[4];
  const float* b2  = (const float*)d_in[5];
  const float* WT1 = (const float*)d_in[6];
  const float* bT1 = (const float*)d_in[7];
  const float* WT2 = (const float*)d_in[8];
  const float* bT2 = (const float*)d_in[9];
  float* out = (float*)d_out;

  float* gates = (float*)d_ws;            // 9*1024*64 floats
  float* hb = gates + 9 * NSS * NHH;      // 1024*256 floats  (total ws: 3.25 MB)

  prep_hb<<<NSS, HID, 0, stream>>>(xc, WT1, bT1, hb);
  gates_kernel<<<NSS, NHH, 0, stream>>>(xc, W1, b1, W2, b2, gates);

  size_t smem = (size_t)(TS * PITCH_H) * sizeof(_Float16) +
                (size_t)(TS * PITCH_V + 9 * TS * NHH + HID + 5 * TS) * sizeof(float);
  scan_kernel<<<NSS / TS, THREADS, smem, stream>>>(x, WT1, WT2, bT2, hb, gates, out);
}
